// SwinTransformerBlock_66829691126268
// MI455X (gfx1250) — matrix operand order repaired and checked by reading, not compiled, not run on hardware
//
#include <hip/hip_runtime.h>
#include <hip/hip_bf16.h>

typedef __attribute__((ext_vector_type(16))) _Float16 v16h;
typedef __attribute__((ext_vector_type(8)))  float    v8f;

#define BATCH 32
#define HW 3136
#define C 128
#define NWIN_TOT 2048           // B * 64 windows
#define NPAD 64                 // window tokens padded 49 -> 64
#define SCALE_Q 0.17677669529663687f  // 1/sqrt(32)

// ---------- WMMA fragment-swizzle helpers (ISA 7.12.2, 16-bit layouts) ----------
// A-matrix 16x32 f16: lanes 0-15 hold K in {0..7}u{16..23}, lanes 16-31 hold {8..15}u{24..31}
__device__ __forceinline__ int fa_slot(int m, int k) {
  int hi   = ((k & 15) >= 8) ? 1 : 0;
  int lane = (m & 15) | (hi << 4);
  int kk   = k - (hi << 3);                       // {0..7} u {16..23}
  int v    = (kk < 8) ? (kk >> 1) : (4 + ((kk - 16) >> 1));
  int e    = (v << 1) | (k & 1);
  return (lane << 4) | e;
}
// B-matrix 32x16 f16: lanes 0-15 hold K in {0..7}u{16..23} of col n, lanes 16-31 hold {8..15}u{24..31}
__device__ __forceinline__ int fb_slot(int k, int n) {
  int lane = (n & 15) | (((k >> 3) & 1) << 4);
  return (lane << 4) | ((k & 7) | ((k >> 4) << 3));
}
__device__ __forceinline__ v16h ld_frag(const _Float16* p, int lane) {
  return *(const v16h*)(p + (lane << 4));         // 2x ds_load_b128
}
__device__ __forceinline__ v8f wmma_f16(v16h a, v16h b, v8f c) {
  return __builtin_amdgcn_wmma_f32_16x16x32_f16(false, a, false, b, (short)0, c, false, false);
}
__device__ __forceinline__ int regid(int y) {     // Swin mask region along one axis
  return (y < 49) ? 0 : ((y < 53) ? 1 : 2);
}

struct __align__(8) h4 { _Float16 x, y, z, w; };

// ================= Kernel 1: LN1 + roll(-3,-3) + window partition -> f16 =================
__global__ __launch_bounds__(256) void k_ln1(const float* __restrict__ x,
                                             const float* __restrict__ g,
                                             const float* __restrict__ bt,
                                             _Float16* __restrict__ xw) {
  const int row  = blockIdx.x * 8 + (threadIdx.x >> 5);   // 0..131071 (window-order, padded)
  const int lane = threadIdx.x & 31;
  const int c0   = lane * 4;
  _Float16* dst  = xw + (size_t)row * C + c0;
  const int n = row & 63;
  if (n >= 49) { *(uint2*)dst = make_uint2(0u, 0u); return; }   // zero pad rows (wave-uniform)
  const int b_ = row >> 6;
  const int bb = b_ >> 6, w = b_ & 63;
  const int wy = w >> 3, wx = w & 7;
  const int iy = n / 7, ix = n % 7;
  const int y  = (wy * 7 + iy + 3) % 56;    // roll by -SHIFT: src = dst + 3 (mod 56)
  const int xx = (wx * 7 + ix + 3) % 56;
  const size_t tok = (size_t)bb * HW + y * 56 + xx;
  const float4 v = *(const float4*)(x + tok * C + c0);
  float s  = v.x + v.y + v.z + v.w;
  float s2 = v.x * v.x + v.y * v.y + v.z * v.z + v.w * v.w;
  for (int o = 16; o > 0; o >>= 1) { s += __shfl_xor(s, o); s2 += __shfl_xor(s2, o); }
  const float mu = s * (1.f / 128.f);
  const float rs = rsqrtf(s2 * (1.f / 128.f) - mu * mu + 1e-5f);
  h4 o;
  o.x = (_Float16)((v.x - mu) * rs * g[c0 + 0] + bt[c0 + 0]);
  o.y = (_Float16)((v.y - mu) * rs * g[c0 + 1] + bt[c0 + 1]);
  o.z = (_Float16)((v.z - mu) * rs * g[c0 + 2] + bt[c0 + 2]);
  o.w = (_Float16)((v.w - mu) * rs * g[c0 + 3] + bt[c0 + 3]);
  *(h4*)dst = o;
}

// ================= Kernel 2: fused window attention block (per window) =================
// QKV gemm (weights streamed in 2 chunks) -> S=QK^T -> masked softmax -> O=PV -> proj
// -> unroll-shift scatter + residual.  Peak LDS 112KB -> 2 workgroups/WGP.
__global__ __launch_bounds__(256) void k_attn(const _Float16* __restrict__ xw,
                                              const float* __restrict__ qkv_w,
                                              const float* __restrict__ qkv_b,
                                              const float* __restrict__ proj_w,
                                              const float* __restrict__ proj_b,
                                              const float* __restrict__ x_in,
                                              float* __restrict__ x1) {
  __shared__ __align__(32) char smem[112 * 1024];
  // phase 0/1 (QKV):   fXW 0..16K | fW chunk 16..64K | fQ 64..80K | fK 80..96K | fV 96..112K
  // phase 2   (S):     S 0..64K (over fXW+fW)        | fQ, fK, fV live
  // phase 3   (smax):  S live, fP 64..96K (over dead fQ+fK), fV live
  // phase 4/5 (PV+proj): fO 0..16K (over dead S) | fPW 16..48K | fP, fV live
  _Float16* fXW = (_Float16*)(smem);               // [4 mi][4 kk][512]        16KB
  _Float16* fW  = (_Float16*)(smem + 16 * 1024);   // [4 kk][12 nt][512]       48KB
  float*    S   = (float*)(smem);                  // [4 h][64][64]            64KB
  _Float16* fQ  = (_Float16*)(smem + 64 * 1024);   // [4 h][4 mi][512]         16KB
  _Float16* fK  = (_Float16*)(smem + 80 * 1024);   // [4 h][4 nt][512]         16KB
  _Float16* fV  = (_Float16*)(smem + 96 * 1024);   // [4 h][2 kc][2 dt][512]   16KB
  _Float16* fP  = (_Float16*)(smem + 64 * 1024);   // [4 h][4 mi][2 kc][512]   32KB
  _Float16* fO  = (_Float16*)(smem);               // [4 mi][4 kk][512]        16KB
  _Float16* fPW = (_Float16*)(smem + 16 * 1024);   // [4 kk][8 nt][512]        32KB

  const int tid = threadIdx.x, wv = tid >> 5, lane = tid & 31;
  const int cl = lane & 15, rh = (lane >> 4) << 3;
  const int b_ = blockIdx.x;
  const size_t R = (size_t)b_ * NPAD;
  v8f zero = {0.f, 0.f, 0.f, 0.f, 0.f, 0.f, 0.f, 0.f};

  // ---- phase 0: stage LN'd window tile (A-frags); prefetch proj weights toward L2 ----
  for (int e = tid; e < 64 * 128; e += 256) {
    int m = e >> 7, k = e & 127;
    fXW[((m >> 4) * 4 + (k >> 5)) * 512 + fa_slot(m & 15, k & 31)] = xw[(R + m) * C + k];
  }
  __builtin_prefetch(proj_w + tid * 64, 0, 1);     // 256 lanes x 256B apart = 64KB region

  // ---- phase 1: QKV GEMM (64x384, K=128), weights streamed in two 192-col chunks ----
  for (int half = 0; half < 2; ++half) {
    for (int e = tid; e < 128 * 192; e += 256) {
      int k = e / 192, n = e % 192;
      fW[((k >> 5) * 12 + (n >> 4)) * 512 + fb_slot(k & 31, n & 15)] =
          (_Float16)qkv_w[k * 384 + half * 192 + n];
    }
    __syncthreads();
    for (int t = wv; t < 48; t += 8) {            // 4 mi x 12 nt per chunk
      int mi = t / 12, ntl = t % 12;
      v8f acc = zero;
      for (int kk = 0; kk < 4; ++kk)
        acc = wmma_f16(ld_frag(fXW + (mi * 4 + kk) * 512, lane),
                       ld_frag(fW + (kk * 12 + ntl) * 512, lane), acc);
      const int nt  = half * 12 + ntl;            // 0..23 (wave-uniform)
      const int sec = nt >> 3;                    // wave-uniform: 0=q 1=k 2=v
      const int col = nt * 16 + cl;               // 0..383
      const float bias = qkv_b[col];
      const int c = col & 127, h = c >> 5, d = c & 31;
      if (sec == 0) {
        for (int r = 0; r < 8; ++r) {
          int m = mi * 16 + r + rh;
          fQ[(h * 4 + (m >> 4)) * 512 + fa_slot(m & 15, d)] = (_Float16)((acc[r] + bias) * SCALE_Q);
        }
      } else if (sec == 1) {                      // K stored transposed (B-frags)
        for (int r = 0; r < 8; ++r) {
          int m = mi * 16 + r + rh;
          fK[(h * 4 + (m >> 4)) * 512 + fb_slot(d, m & 15)] = (_Float16)(acc[r] + bias);
        }
      } else {
        for (int r = 0; r < 8; ++r) {
          int m = mi * 16 + r + rh;
          fV[((h * 2 + (m >> 5)) * 2 + (d >> 4)) * 512 + fb_slot(m & 31, d & 15)] =
              (_Float16)(acc[r] + bias);
        }
      }
    }
    __syncthreads();                              // protect fW restage / fXW before S overlay
  }

  // ---- phase 2: S = Q K^T per head (4x [64x64], K=32) ----
  for (int t = wv; t < 64; t += 8) {
    int h = t >> 4, i = (t >> 2) & 3, j = t & 3;
    v8f acc = wmma_f16(ld_frag(fQ + (h * 4 + i) * 512, lane),
                       ld_frag(fK + (h * 4 + j) * 512, lane), zero);
    float* Sh = S + h * 64 * 64;
    for (int r = 0; r < 8; ++r)
      Sh[(i * 16 + r + rh) * 64 + j * 16 + cl] = acc[r];
  }
  __syncthreads();

  // ---- phase 3: masked softmax (mask recomputed from shift-region ids), P -> A-frags ----
  {
    int h = tid >> 6, i = tid & 63;
    const int w = b_ & 63, wy = w >> 3, wx = w & 7;
    int rq = regid(wy * 7 + i / 7) * 3 + regid(wx * 7 + i % 7);
    float* Sr = S + (h * 64 + i) * 64;
    float mx = -1e30f;
    for (int j = 0; j < 49; ++j) {
      int rk = regid(wy * 7 + j / 7) * 3 + regid(wx * 7 + j % 7);
      float v = Sr[j] + ((rk == rq) ? 0.f : -100.f);
      mx = fmaxf(mx, v);
    }
    float sum = 0.f;
    for (int j = 0; j < 64; ++j) {
      float p = 0.f;
      if (j < 49) {
        int rk = regid(wy * 7 + j / 7) * 3 + regid(wx * 7 + j % 7);
        float v = Sr[j] + ((rk == rq) ? 0.f : -100.f);
        p = __expf(v - mx);
        sum += p;
      }
      Sr[j] = p;
    }
    float inv = 1.f / sum;
    for (int j = 0; j < 64; ++j)
      fP[((h * 4 + (i >> 4)) * 2 + (j >> 5)) * 512 + fa_slot(i & 15, j & 31)] = (_Float16)(Sr[j] * inv);
  }
  __syncthreads();

  // ---- phase 4: O = P V (per head 64x32, K=64) into A-frags for proj; stage proj_w ----
  for (int t = wv; t < 32; t += 8) {
    int h = t >> 3, i = (t >> 1) & 3, dt = t & 1;
    v8f acc = wmma_f16(ld_frag(fP + ((h * 4 + i) * 2 + 0) * 512, lane),
                       ld_frag(fV + ((h * 2 + 0) * 2 + dt) * 512, lane), zero);
    acc = wmma_f16(ld_frag(fP + ((h * 4 + i) * 2 + 1) * 512, lane),
                   ld_frag(fV + ((h * 2 + 1) * 2 + dt) * 512, lane), acc);
    int c = h * 32 + dt * 16 + cl;          // output channel 0..127
    for (int r = 0; r < 8; ++r) {
      int m = i * 16 + r + rh;
      fO[(i * 4 + (c >> 5)) * 512 + fa_slot(m & 15, c & 31)] = (_Float16)acc[r];
    }
  }
  for (int e = tid; e < 128 * 128; e += 256) {
    int k = e >> 7, n = e & 127;
    fPW[((k >> 5) * 8 + (n >> 4)) * 512 + fb_slot(k & 31, n & 15)] = (_Float16)proj_w[e];
  }
  __syncthreads();

  // ---- phase 5: proj GEMM (64x128, K=128) -> roll(+3,+3) scatter + residual into x1 ----
  for (int t = wv; t < 32; t += 8) {
    int mi = t >> 3, nt = t & 7;
    v8f acc = zero;
    for (int kk = 0; kk < 4; ++kk)
      acc = wmma_f16(ld_frag(fO + (mi * 4 + kk) * 512, lane),
                     ld_frag(fPW + (kk * 8 + nt) * 512, lane), acc);
    int c = nt * 16 + cl;
    float bias = proj_b[c];
    const int w = b_ & 63, bb = b_ >> 6, wy = w >> 3, wx = w & 7;
    for (int r = 0; r < 8; ++r) {
      int n = mi * 16 + r + rh;
      if (n < 49) {
        int y  = (wy * 7 + n / 7 + 3) % 56;
        int xx = (wx * 7 + n % 7 + 3) % 56;
        size_t tok = (size_t)bb * HW + y * 56 + xx;
        x1[tok * C + c] = acc[r] + bias + x_in[tok * C + c];
      }
    }
  }
}

// ================= Kernel 3: LN2 -> f16 =================
__global__ __launch_bounds__(256) void k_ln2(const float* __restrict__ x1,
                                             const float* __restrict__ g,
                                             const float* __restrict__ bt,
                                             _Float16* __restrict__ out) {
  const size_t row = (size_t)blockIdx.x * 8 + (threadIdx.x >> 5);
  const int lane = threadIdx.x & 31, c0 = lane * 4;
  const float4 v = *(const float4*)(x1 + row * C + c0);
  float s  = v.x + v.y + v.z + v.w;
  float s2 = v.x * v.x + v.y * v.y + v.z * v.z + v.w * v.w;
  for (int o = 16; o > 0; o >>= 1) { s += __shfl_xor(s, o); s2 += __shfl_xor(s2, o); }
  const float mu = s * (1.f / 128.f);
  const float rs = rsqrtf(s2 * (1.f / 128.f) - mu * mu + 1e-5f);
  h4 o;
  o.x = (_Float16)((v.x - mu) * rs * g[c0 + 0] + bt[c0 + 0]);
  o.y = (_Float16)((v.y - mu) * rs * g[c0 + 1] + bt[c0 + 1]);
  o.z = (_Float16)((v.z - mu) * rs * g[c0 + 2] + bt[c0 + 2]);
  o.w = (_Float16)((v.w - mu) * rs * g[c0 + 3] + bt[c0 + 3]);
  *(h4*)(out + row * C + c0) = o;
}

// ================= Kernel 4: fused MLP: GEMM1+GELU (h in LDS frags) -> GEMM2 + residual =================
__global__ __launch_bounds__(256) void k_mlp(const _Float16* __restrict__ xln2,
                                             const float* __restrict__ w1,
                                             const float* __restrict__ b1,
                                             const float* __restrict__ w2,
                                             const float* __restrict__ b2,
                                             const float* __restrict__ x1,
                                             float* __restrict__ out) {
  __shared__ __align__(32) char smem[96 * 1024];
  _Float16* fA  = (_Float16*)(smem);               // [4 mi][4 kk][512]   16KB
  _Float16* fWk = (_Float16*)(smem + 16 * 1024);   // W1 chunk [4][4][512] / W2 chunk [8][512]
  _Float16* fH  = (_Float16*)(smem + 32 * 1024);   // [4 mi][16 kc][512]  64KB (gelu'd hidden, A-frags)
  const int tid = threadIdx.x, wv = tid >> 5, lane = tid & 31;
  const int cl = lane & 15, rh = (lane >> 4) << 3;
  const size_t m0 = (size_t)blockIdx.x * 64;
  v8f zero = {0.f, 0.f, 0.f, 0.f, 0.f, 0.f, 0.f, 0.f};

  for (int e = tid; e < 64 * 128; e += 256) {
    int m = e >> 7, k = e & 127;
    fA[((m >> 4) * 4 + (k >> 5)) * 512 + fa_slot(m & 15, k & 31)] = xln2[(m0 + m) * C + k];
  }
  __builtin_prefetch(w2 + tid * 256, 0, 1);        // 256 lanes x 1KB apart = 256KB region
  __syncthreads();

  // GEMM1: [64,128] x [128,512] in 8 column chunks of 64; bias + exact GELU -> fH
  for (int c0 = 0; c0 < 8; ++c0) {
    for (int e = tid; e < 128 * 64; e += 256) {
      int k = e >> 6, n = e & 63;
      fWk[((k >> 5) * 4 + (n >> 4)) * 512 + fb_slot(k & 31, n & 15)] = (_Float16)w1[k * 512 + c0 * 64 + n];
    }
    __syncthreads();
    for (int t = wv; t < 16; t += 8) {
      int mi = t >> 2, nt = t & 3;
      v8f acc = zero;
      for (int kk = 0; kk < 4; ++kk)
        acc = wmma_f16(ld_frag(fA + (mi * 4 + kk) * 512, lane),
                       ld_frag(fWk + (kk * 4 + nt) * 512, lane), acc);
      int hc = c0 * 64 + nt * 16 + cl;      // hidden col 0..511
      float bias = b1[hc];
      for (int r = 0; r < 8; ++r) {
        float v = acc[r] + bias;
        float ge = 0.5f * v * (1.0f + erff(v * 0.7071067811865476f));
        int m = mi * 16 + r + rh;
        fH[(mi * 16 + (hc >> 5)) * 512 + fa_slot(m & 15, hc & 31)] = (_Float16)ge;
      }
    }
    __syncthreads();
  }

  // GEMM2: [64,512] x [512,128] streamed in 16 K-chunks of 32
  v8f accs[4] = {zero, zero, zero, zero};
  const int mi = wv & 3, ng = wv >> 2;
  for (int kc = 0; kc < 16; ++kc) {
    for (int e = tid; e < 32 * 128; e += 256) {
      int k = e >> 7, n = e & 127;
      fWk[(n >> 4) * 512 + fb_slot(k & 31, n & 15)] = (_Float16)w2[(kc * 32 + k) * 128 + n];
    }
    __syncthreads();
    v16h a = ld_frag(fH + (mi * 16 + kc) * 512, lane);
    for (int t = 0; t < 4; ++t)
      accs[t] = wmma_f16(a, ld_frag(fWk + (ng * 4 + t) * 512, lane), accs[t]);
    __syncthreads();
  }
  for (int t = 0; t < 4; ++t) {
    int c = (ng * 4 + t) * 16 + cl;
    float bias = b2[c];
    for (int r = 0; r < 8; ++r) {
      size_t m = m0 + mi * 16 + r + rh;
      out[m * C + c] = accs[t][r] + bias + x1[m * C + c];
    }
  }
}

// ================= host launcher =================
extern "C" void kernel_launch(void* const* d_in, const int* in_sizes, int n_in,
                              void* d_out, int out_size, void* d_ws, size_t ws_size,
                              hipStream_t stream) {
  const float* x      = (const float*)d_in[0];
  const float* ln1_g  = (const float*)d_in[1];
  const float* ln1_b  = (const float*)d_in[2];
  const float* qkv_w  = (const float*)d_in[3];
  const float* qkv_b  = (const float*)d_in[4];
  const float* proj_w = (const float*)d_in[5];
  const float* proj_b = (const float*)d_in[6];
  const float* ln2_g  = (const float*)d_in[7];
  const float* ln2_b  = (const float*)d_in[8];
  const float* mlp_w1 = (const float*)d_in[9];
  const float* mlp_b1 = (const float*)d_in[10];
  const float* mlp_w2 = (const float*)d_in[11];
  const float* mlp_b2 = (const float*)d_in[12];
  float* outp = (float*)d_out;

  // workspace layout: x1 f32 (51.4MB) | xw f16 (33.6MB, reused as xln2)
  float*    x1   = (float*)d_ws;
  _Float16* xw   = (_Float16*)((char*)d_ws + (size_t)BATCH * HW * C * 4);
  _Float16* xln2 = xw;  // xw dead after k_attn

  k_ln1 <<<NWIN_TOT * NPAD / 8, 256, 0, stream>>>(x, ln1_g, ln1_b, xw);
  k_attn<<<NWIN_TOT,            256, 0, stream>>>(xw, qkv_w, qkv_b, proj_w, proj_b, x, x1);
  k_ln2 <<<BATCH * HW / 8,      256, 0, stream>>>(x1, ln2_g, ln2_b, xln2);
  k_mlp <<<BATCH * HW / 64,     256, 0, stream>>>(xln2, mlp_w1, mlp_b1, mlp_w2, mlp_b2, x1, outp);
  (void)in_sizes; (void)n_in; (void)out_size; (void)ws_size;
}
